// IGNet_14602888806924
// MI455X (gfx1250) — matrix operand order repaired and checked by reading, not compiled, not run on hardware
//
#include <hip/hip_runtime.h>
#include <hip/hip_bf16.h>
#include <math.h>

#define DD 128

typedef __attribute__((ext_vector_type(16))) __bf16         v16bf;
typedef __attribute__((ext_vector_type(8)))  __bf16         v8bf;
typedef __attribute__((ext_vector_type(8)))  float          v8f;
typedef __attribute__((ext_vector_type(8)))  unsigned short v8u16;
typedef __attribute__((ext_vector_type(16))) unsigned short v16u16;

__device__ __forceinline__ unsigned short f32_to_bf16_rne(float f) {
  union { float f; unsigned int u; } v; v.f = f;
  unsigned int u = v.u;
  u += 0x7FFFu + ((u >> 16) & 1u);   // round-to-nearest-even
  return (unsigned short)(u >> 16);
}

// ---------------- elementwise helpers ----------------

__global__ void k_zero(float* __restrict__ p, long long n) {
  long long i = (long long)blockIdx.x * blockDim.x + threadIdx.x;
  if (i < n) p[i] = 0.0f;
}

__global__ void k_f32_to_bf16(const float* __restrict__ s,
                              unsigned short* __restrict__ d, long long n) {
  long long i = (long long)blockIdx.x * blockDim.x + threadIdx.x;
  if (i < n) d[i] = f32_to_bf16_rne(s[i]);
}

// s[dst[e]][4q..4q+3] += h[src[e]][4q..4q+3]; optionally deg[dst[e]] += 1.
// One thread per 4 contiguous floats => each wave handles exactly one edge,
// so src/dst index loads are wave-uniform.
__global__ void k_scatter_add4(const float* __restrict__ h, const int* __restrict__ src,
                               const int* __restrict__ dst, float* __restrict__ s,
                               float* __restrict__ deg, long long total4) {
  long long i = (long long)blockIdx.x * blockDim.x + threadIdx.x;
  if (i < total4) {
    int e = (int)(i >> 5);           // 32 quads per edge row of 128
    int q = (int)(i & 31);
    int sr = src[e], dt = dst[e];
    const float4 v = *(const float4*)(h + (size_t)sr * DD + q * 4);
    float* o = s + (size_t)dt * DD + q * 4;
    atomicAdd(o + 0, v.x);
    atomicAdd(o + 1, v.y);
    atomicAdd(o + 2, v.z);
    atomicAdd(o + 3, v.w);
    if (deg != nullptr && q == 0) atomicAdd(&deg[dt], 1.0f);
  }
}

// h_neigh = (deg>0 ? s/deg : 0), converted to bf16
__global__ void k_mean_bf16(const float* __restrict__ s, const float* __restrict__ deg,
                            unsigned short* __restrict__ outbf, long long total) {
  long long i = (long long)blockIdx.x * blockDim.x + threadIdx.x;
  if (i < total) {
    float dg = deg[i >> 7];
    float v  = (dg > 0.0f) ? s[i] / dg : 0.0f;
    outbf[i] = f32_to_bf16_rne(v);
  }
}

// res = x + h1 + 0.5*h2 ; write fp32 in-place over h2 and bf16 copy
__global__ void k_res(const float* __restrict__ x, const float* __restrict__ h1,
                      float* __restrict__ h2res, unsigned short* __restrict__ bfres,
                      long long total) {
  long long i = (long long)blockIdx.x * blockDim.x + threadIdx.x;
  if (i < total) {
    float r = x[i] + h1[i] + 0.5f * h2res[i];
    h2res[i] = r;
    bfres[i] = f32_to_bf16_rne(r);
  }
}

// out = softmax(scores/N) blend of res and view2
__global__ void k_final(const float* __restrict__ res, const float* __restrict__ view2,
                        const float* __restrict__ scores, float invN,
                        float* __restrict__ out, long long total) {
  long long i = (long long)blockIdx.x * blockDim.x + threadIdx.x;
  if (i < total) {
    float s0 = scores[0] * invN, s1 = scores[1] * invN;
    float m  = fmaxf(s0, s1);
    float e0 = __expf(s0 - m), e1 = __expf(s1 - m);
    float inv = 1.0f / (e0 + e1);
    out[i] = (e0 * res[i] + e1 * view2[i]) * inv;
  }
}

// ---------------- WMMA fragment loaders (this code's contiguous k order, same for A and B) ----------------
//
// A (16x32 bf16, M x K): lane L, m = L&15, half = L>>4.
//   VGPR 0..3 hold K = half*16 + {0..7};  VGPR 4..7 hold K = half*16 + 8 + {0..7}.
//   => two contiguous 8-element (16 B) loads per lane per K-step of 32.
// B (32x16 bf16, K x N) with B = W^T: lane L, n = L&15, half = L>>4,
//   holds K = half*16 + {0..15} contiguous => one 32 B load from row n of W.

__device__ __forceinline__ v16bf load_a_frag(const unsigned short* arow,
                                             int kk, int half) {
  v8bf lo = __builtin_bit_cast(v8bf, *(const v8u16*)(arow + kk * 32 + half * 16));
  v8bf hi = __builtin_bit_cast(v8bf, *(const v8u16*)(arow + kk * 32 + half * 16 + 8));
  v16bf a;
#pragma unroll
  for (int i = 0; i < 8; ++i) { a[i] = lo[i]; a[8 + i] = hi[i]; }
  return a;
}

__device__ __forceinline__ v16bf load_b_frag(const unsigned short* __restrict__ wrow,
                                             int kk, int half) {
  return __builtin_bit_cast(v16bf, *(const v16u16*)(wrow + kk * 32 + half * 16));
}

// C = A1*W1^T + A2*W2^T + bias.  Each wave computes TWO 16x16 output tiles
// (32 consecutive rows, same 16 cols) so each weight fragment feeds 2 WMMAs.
// Optionally also writes a bf16 copy of C (safe to alias A1: each block only
// touches its own row slice, and stores are data-dependent on all loads).
__global__ __launch_bounds__(256) void k_gemm_dual(
    const unsigned short* A1, const unsigned short* __restrict__ W1,
    const unsigned short* __restrict__ A2, const unsigned short* __restrict__ W2,
    const float* __restrict__ bias, float* __restrict__ Cout,
    unsigned short* Cbf, int nrows) {
  const int lane = threadIdx.x & 31;
  const int wave = threadIdx.x >> 5;
  const int half = lane >> 4;
  const int l16  = lane & 15;
  const int row0 = blockIdx.x * 32;
  const int col0 = wave * 16;

  v8f acc0, acc1;
  const float b = bias[col0 + l16];
#pragma unroll
  for (int i = 0; i < 8; ++i) { acc0[i] = b; acc1[i] = b; }

  int ra = row0 + l16;        if (ra >= nrows) ra = nrows - 1;  // keep EXEC all-ones
  int rb = row0 + 16 + l16;   if (rb >= nrows) rb = nrows - 1;
  const unsigned short* a1a = A1 + (size_t)ra * DD;
  const unsigned short* a1b = A1 + (size_t)rb * DD;
  const unsigned short* a2a = A2 + (size_t)ra * DD;
  const unsigned short* a2b = A2 + (size_t)rb * DD;
  const unsigned short* w1  = W1 + (size_t)(col0 + l16) * DD;
  const unsigned short* w2  = W2 + (size_t)(col0 + l16) * DD;

#pragma unroll
  for (int kk = 0; kk < 4; ++kk) {
    v16bf bf = load_b_frag(w1, kk, half);
    v16bf fa = load_a_frag(a1a, kk, half);
    v16bf fb = load_a_frag(a1b, kk, half);
    acc0 = __builtin_amdgcn_wmma_f32_16x16x32_bf16(false, fa, false, bf,
                                                   (short)0, acc0, false, false);
    acc1 = __builtin_amdgcn_wmma_f32_16x16x32_bf16(false, fb, false, bf,
                                                   (short)0, acc1, false, false);
  }
#pragma unroll
  for (int kk = 0; kk < 4; ++kk) {
    v16bf bf = load_b_frag(w2, kk, half);
    v16bf fa = load_a_frag(a2a, kk, half);
    v16bf fb = load_a_frag(a2b, kk, half);
    acc0 = __builtin_amdgcn_wmma_f32_16x16x32_bf16(false, fa, false, bf,
                                                   (short)0, acc0, false, false);
    acc1 = __builtin_amdgcn_wmma_f32_16x16x32_bf16(false, fb, false, bf,
                                                   (short)0, acc1, false, false);
  }

  // C/D layout: VGPR v, lanes 0-15 -> M=v, lanes 16-31 -> M=v+8; N = lane&15
#pragma unroll
  for (int v = 0; v < 8; ++v) {
    int r0 = row0 + v + half * 8;
    int r1 = r0 + 16;
    int c  = col0 + l16;
    if (r0 < nrows) {
      Cout[(size_t)r0 * DD + c] = acc0[v];
      if (Cbf != nullptr) Cbf[(size_t)r0 * DD + c] = f32_to_bf16_rne(acc0[v]);
    }
    if (r1 < nrows) {
      Cout[(size_t)r1 * DD + c] = acc1[v];
      if (Cbf != nullptr) Cbf[(size_t)r1 * DD + c] = f32_to_bf16_rne(acc1[v]);
    }
  }
}

// scores[view] += sum over tile of att[col] * tanh(A*W_attn^T + b_attn)
__global__ __launch_bounds__(256) void k_gemm_attn(
    const unsigned short* __restrict__ Ares, const unsigned short* __restrict__ Aview2,
    const unsigned short* __restrict__ W, const float* __restrict__ bias,
    const float* __restrict__ att, float* __restrict__ scores, int nrows) {
  const unsigned short* A = blockIdx.y ? Aview2 : Ares;
  const int lane = threadIdx.x & 31;
  const int wave = threadIdx.x >> 5;
  const int half = lane >> 4;
  const int l16  = lane & 15;
  const int row0 = blockIdx.x * 16;
  const int col0 = wave * 16;

  v8f acc;
  const float b = bias[col0 + l16];
#pragma unroll
  for (int i = 0; i < 8; ++i) acc[i] = b;

  int arow = row0 + l16;
  if (arow >= nrows) arow = nrows - 1;
  const unsigned short* ar = A + (size_t)arow * DD;
  const unsigned short* wr = W + (size_t)(col0 + l16) * DD;

#pragma unroll
  for (int kk = 0; kk < 4; ++kk) {
    v16bf af = load_a_frag(ar, kk, half);
    v16bf bf = load_b_frag(wr, kk, half);
    acc = __builtin_amdgcn_wmma_f32_16x16x32_bf16(false, af, false, bf,
                                                  (short)0, acc, false, false);
  }

  const float av = att[col0 + l16];
  float part = 0.0f;
#pragma unroll
  for (int v = 0; v < 8; ++v) {
    int r = row0 + v + half * 8;
    if (r < nrows) part += tanhf(acc[v]);
  }
  part *= av;
#pragma unroll
  for (int off = 16; off >= 1; off >>= 1) part += __shfl_xor(part, off, 32);
  if (lane == 0) atomicAdd(&scores[blockIdx.y], part);
}

// ---------------- host launch ----------------

extern "C" void kernel_launch(void* const* d_in, const int* in_sizes, int n_in,
                              void* d_out, int out_size, void* d_ws, size_t ws_size,
                              hipStream_t stream) {
  (void)n_in; (void)out_size; (void)ws_size;
  const float* x        = (const float*)d_in[0];
  const float* view2    = (const float*)d_in[1];
  const float* W_neigh1 = (const float*)d_in[2];
  const float* W_self1  = (const float*)d_in[3];
  const float* b_self1  = (const float*)d_in[4];
  const float* W_neigh2 = (const float*)d_in[5];
  const float* W_self2  = (const float*)d_in[6];
  const float* b_self2  = (const float*)d_in[7];
  const float* W_attn   = (const float*)d_in[8];
  const float* b_attn   = (const float*)d_in[9];
  const float* att      = (const float*)d_in[10];
  const int*   src      = (const int*)d_in[11];
  const int*   dst      = (const int*)d_in[12];
  float* out = (float*)d_out;

  const long long N   = in_sizes[0] / DD;
  const int       E   = in_sizes[11];
  const long long ND  = N * (long long)DD;
  const long long EQ4 = (long long)E * (DD / 4);   // edge quads
  const long long WW  = (long long)DD * DD;

  char* p = (char*)d_ws;
  auto take = [&](size_t bytes) {
    char* r = p; p += (bytes + 255) & ~(size_t)255; return r;
  };
  float*          s_buf  = (float*)take((size_t)ND * 4);
  float*          deg    = (float*)take((size_t)N * 4);
  float*          h1     = (float*)take((size_t)ND * 4);
  float*          h2res  = (float*)take((size_t)ND * 4);
  unsigned short* bfA    = (unsigned short*)take((size_t)ND * 2);
  unsigned short* bfB    = (unsigned short*)take((size_t)ND * 2);
  unsigned short* bfW    = (unsigned short*)take((size_t)(5 * WW) * 2);
  float*          scores = (float*)take(256);

  const int TB = 256;
  auto nb = [](long long n) { return (unsigned)((n + 255) / 256); };
  const unsigned tiles16 = (unsigned)((N + 15) / 16);
  const unsigned tiles32 = (unsigned)((N + 31) / 32);

  // init accumulators (workspace is poisoned before timing)
  k_zero<<<nb(ND), TB, 0, stream>>>(s_buf, ND);
  k_zero<<<nb(N),  TB, 0, stream>>>(deg, N);
  k_zero<<<1, 64,  0, stream>>>(scores, 64);

  // weights + x to bf16
  k_f32_to_bf16<<<nb(WW), TB, 0, stream>>>(W_self1,  bfW + 0 * WW, WW);
  k_f32_to_bf16<<<nb(WW), TB, 0, stream>>>(W_neigh1, bfW + 1 * WW, WW);
  k_f32_to_bf16<<<nb(WW), TB, 0, stream>>>(W_self2,  bfW + 2 * WW, WW);
  k_f32_to_bf16<<<nb(WW), TB, 0, stream>>>(W_neigh2, bfW + 3 * WW, WW);
  k_f32_to_bf16<<<nb(WW), TB, 0, stream>>>(W_attn,   bfW + 4 * WW, WW);
  k_f32_to_bf16<<<nb(ND), TB, 0, stream>>>(x, bfA, ND);

  // ---- SAGE layer 1 (degree fused into first scatter) ----
  k_scatter_add4<<<nb(EQ4), TB, 0, stream>>>(x, src, dst, s_buf, deg, EQ4);
  k_mean_bf16<<<nb(ND), TB, 0, stream>>>(s_buf, deg, bfB, ND);
  // writes h1 (fp32) and bf16(h1) over bfA in the same pass
  k_gemm_dual<<<dim3(tiles32), TB, 0, stream>>>(bfA, bfW + 0 * WW,
                                                bfB, bfW + 1 * WW,
                                                b_self1, h1, bfA, (int)N);

  // ---- SAGE layer 2 ----
  k_zero<<<nb(ND), TB, 0, stream>>>(s_buf, ND);
  k_scatter_add4<<<nb(EQ4), TB, 0, stream>>>(h1, src, dst, s_buf, nullptr, EQ4);
  k_mean_bf16<<<nb(ND), TB, 0, stream>>>(s_buf, deg, bfB, ND);
  k_gemm_dual<<<dim3(tiles32), TB, 0, stream>>>(bfA, bfW + 2 * WW,
                                                bfB, bfW + 3 * WW,
                                                b_self2, h2res, nullptr, (int)N);

  // ---- residual + attention fusion ----
  k_res<<<nb(ND), TB, 0, stream>>>(x, h1, h2res, bfA, ND);   // res in h2res, bf16 in bfA
  k_f32_to_bf16<<<nb(ND), TB, 0, stream>>>(view2, bfB, ND);
  k_gemm_attn<<<dim3(tiles16, 2), TB, 0, stream>>>(bfA, bfB, bfW + 4 * WW,
                                                   b_attn, att, scores, (int)N);
  k_final<<<nb(ND), TB, 0, stream>>>(h2res, view2, scores,
                                     1.0f / (float)N, out, ND);
}
